// Multi_CrossAttention_39453569581285
// MI455X (gfx1250) — matrix operand order repaired and checked by reading, not compiled, not run on hardware
//
#include <hip/hip_runtime.h>

#define DEVINL __device__ __forceinline__

typedef __attribute__((ext_vector_type(16))) __bf16 v16bf;
typedef __attribute__((ext_vector_type(8)))  __bf16 v8bf;
typedef __attribute__((ext_vector_type(8)))  float  v8f;

constexpr int Bn  = 4;
constexpr int Sn  = 2048;
constexpr int HID = 1024;
constexpr int Hh  = 16;
constexpr int DHn = 64;
constexpr int Mrows = Bn * Sn;          // 8192
constexpr float SCL  = 0.125f;          // 1/sqrt(64)
constexpr float NEGS = -1.0e9f * SCL;   // masked value after scaling

DEVINL __bf16 f2bf(float f) {
  union { float f; unsigned u; } v; v.f = f;
  unsigned r = v.u + 0x7FFFu + ((v.u >> 16) & 1u);   // round-to-nearest-even
  unsigned short h = (unsigned short)(r >> 16);
  return __builtin_bit_cast(__bf16, h);
}

DEVINL v8f wmma_bf16(v16bf a, v16bf b, v8f c) {
  // D = A(16x32 bf16) * B(32x16 bf16) + C(16x16 f32)
  return __builtin_amdgcn_wmma_f32_16x16x32_bf16(
      false, a, false, b, (short)0, c, false, false);
}

// A-fragment (16x32, bf16, row-major source, lda elements): row = lane&15,
// K = {base..base+7} U {base+16..base+23}, base = (lane>>4)*8.
DEVINL v16bf ldA(const __bf16* A, int lda, int m0, int kc, int lane) {
  int m = m0 + (lane & 15);
  int base = (lane >> 4) * 8;
  const __bf16* p = A + (size_t)m * lda + kc + base;
  v8bf lo = *(const v8bf*)(p);
  v8bf hi = *(const v8bf*)(p + 16);
  return __builtin_shufflevector(lo, hi, 0,1,2,3,4,5,6,7,8,9,10,11,12,13,14,15);
}

// B-fragment (32x16, bf16) from BT[n][k] row-major (ldb elements):
// col n = n0 + lane&15, K = {base..base+7} U {base+16..base+23}, base = kc + (lane>>4)*8.
DEVINL v16bf ldB(const __bf16* BT, int ldb, int n0, int kc, int lane) {
  int n = n0 + (lane & 15);
  int k = kc + (lane >> 4) * 8;
  const __bf16* p = BT + (size_t)n * ldb + k;
  v8bf lo = *(const v8bf*)(p);
  v8bf hi = *(const v8bf*)(p + 16);
  return __builtin_shufflevector(lo, hi, 0,1,2,3,4,5,6,7,8,9,10,11,12,13,14,15);
}

// Double-buffered 32x64 GEMM tile: C[8] = A[m0..m0+31][:] * BT^T[:, n0..n0+63]
DEVINL void gemm32x64(const __bf16* __restrict__ A, const __bf16* __restrict__ BT,
                      int m0, int n0, int lane, v8f c[8]) {
  v16bf a[2][2], b[2][4];
  a[0][0] = ldA(A, HID, m0, 0, lane);
  a[0][1] = ldA(A, HID, m0 + 16, 0, lane);
#pragma unroll
  for (int i = 0; i < 4; ++i) b[0][i] = ldB(BT, HID, n0 + i * 16, 0, lane);

  for (int kc = 0; kc < HID; kc += 64) {
    // issue loads for k = kc+32 (buffer 1), then compute k = kc (buffer 0)
    a[1][0] = ldA(A, HID, m0, kc + 32, lane);
    a[1][1] = ldA(A, HID, m0 + 16, kc + 32, lane);
#pragma unroll
    for (int i = 0; i < 4; ++i) b[1][i] = ldB(BT, HID, n0 + i * 16, kc + 32, lane);
#pragma unroll
    for (int r = 0; r < 2; ++r)
#pragma unroll
      for (int i = 0; i < 4; ++i)
        c[r * 4 + i] = wmma_bf16(a[0][r], b[0][i], c[r * 4 + i]);

    // issue loads for k = kc+64 (buffer 0), then compute k = kc+32 (buffer 1)
    if (kc + 64 < HID) {
      a[0][0] = ldA(A, HID, m0, kc + 64, lane);
      a[0][1] = ldA(A, HID, m0 + 16, kc + 64, lane);
#pragma unroll
      for (int i = 0; i < 4; ++i) b[0][i] = ldB(BT, HID, n0 + i * 16, kc + 64, lane);
    }
#pragma unroll
    for (int r = 0; r < 2; ++r)
#pragma unroll
      for (int i = 0; i < 4; ++i)
        c[r * 4 + i] = wmma_bf16(a[1][r], b[1][i], c[r * 4 + i]);
  }
}

// ---------------- conversion kernels ----------------

// 8 elements per thread: float4 x2 in, one 16B bf16 vector out
__global__ void cvt_copy8(const float4* __restrict__ in, v8bf* __restrict__ out, int n8) {
  int i = blockIdx.x * blockDim.x + threadIdx.x;
  if (i < n8) {
    float4 f0 = in[2 * i], f1 = in[2 * i + 1];
    v8bf o;
    o[0] = f2bf(f0.x); o[1] = f2bf(f0.y); o[2] = f2bf(f0.z); o[3] = f2bf(f0.w);
    o[4] = f2bf(f1.x); o[5] = f2bf(f1.y); o[6] = f2bf(f1.z); o[7] = f2bf(f1.w);
    out[i] = o;
  }
}

// LDS-tiled transpose: out[n*rows + k] = bf16(in[k*cols + n]); coalesced both sides
__global__ void cvt_tr(const float* __restrict__ in, __bf16* __restrict__ out,
                       int rows, int cols) {
  __shared__ __bf16 tile[32][33];
  int c0 = blockIdx.x * 32, r0 = blockIdx.y * 32;
  int tx = threadIdx.x & 31, ty = threadIdx.x >> 5;   // 256 thr: ty = 0..7
#pragma unroll
  for (int rr = ty; rr < 32; rr += 8)
    tile[rr][tx] = f2bf(in[(size_t)(r0 + rr) * cols + c0 + tx]);
  __syncthreads();
#pragma unroll
  for (int rr = ty; rr < 32; rr += 8)
    out[(size_t)(c0 + rr) * rows + r0 + tx] = tile[tx][rr];
}

// ---------------- fused QKV projection ----------------
// waves: 3 gemms * (8192/32 row-tiles) * (1024/64 col-groups) = 12288
__global__ void proj_qkv(const __bf16* __restrict__ xbf, const __bf16* __restrict__ ybf,
                         const __bf16* __restrict__ WqT, const __bf16* __restrict__ WkT,
                         const __bf16* __restrict__ WvT,
                         __bf16* __restrict__ Qo, __bf16* __restrict__ Ko,
                         __bf16* __restrict__ VTo) {
  int wid  = (blockIdx.x * blockDim.x + threadIdx.x) >> 5;
  int lane = threadIdx.x & 31;
  constexpr int TILES = (Mrows / 32) * (HID / 64);   // 4096
  int which = wid / TILES;                           // 0=Q 1=K 2=V
  int t     = wid % TILES;
  int m0 = (t / (HID / 64)) * 32;
  int n0 = (t % (HID / 64)) * 64;
  const __bf16* A  = (which == 2) ? ybf : xbf;
  const __bf16* BT = (which == 0) ? WqT : (which == 1) ? WkT : WvT;

  v8f acc[8] = {};
  gemm32x64(A, BT, m0, n0, lane, acc);

  int lh = lane >> 4, ln = lane & 15;
  if (which < 2) {
    __bf16* dst = (which == 0) ? Qo : Ko;   // [B,H,S,Dh]
#pragma unroll
    for (int r = 0; r < 2; ++r)
#pragma unroll
      for (int i = 0; i < 4; ++i) {
        int n = n0 + i * 16 + ln;
        int h = n >> 6, d = n & 63;
#pragma unroll
        for (int j = 0; j < 8; ++j) {
          int m = m0 + r * 16 + lh * 8 + j;
          int b = m >> 11, s = m & (Sn - 1);
          dst[(((size_t)(b * Hh + h) * Sn + s) * DHn) + d] = f2bf(acc[r * 4 + i][j]);
        }
      }
  } else {
    // V stored transposed: VT[B,H,Dh,S]; 8 consecutive rows m -> contiguous s
#pragma unroll
    for (int r = 0; r < 2; ++r)
#pragma unroll
      for (int i = 0; i < 4; ++i) {
        int n = n0 + i * 16 + ln;
        int h = n >> 6, d = n & 63;
        int m = m0 + r * 16 + lh * 8;
        int b = m >> 11, s = m & (Sn - 1);
        v8bf pk;
#pragma unroll
        for (int j = 0; j < 8; ++j) pk[j] = f2bf(acc[r * 4 + i][j]);
        *(v8bf*)(VTo + ((size_t)(b * Hh + h) * DHn + d) * Sn + s) = pk;
      }
  }
}

// ---------------- flash attention ----------------
// waves: B*H*(S/32) = 4096; one wave owns a 32-row q-tile of one (b,h).
__global__ void attn(const __bf16* __restrict__ Qb, const __bf16* __restrict__ Kb,
                     const __bf16* __restrict__ VTb, const int* __restrict__ maskp,
                     __bf16* __restrict__ ctx) {
  __shared__ __align__(32) __bf16 pTile[8][32 * 32];
  int widb = threadIdx.x >> 5;
  int wid  = blockIdx.x * (blockDim.x >> 5) + widb;
  int lane = threadIdx.x & 31;
  int lh = lane >> 4, ln = lane & 15;

  int qt = wid & 63;             // q-tile (32 rows) within (b,h)
  int bh = wid >> 6;
  int b  = bh >> 4;
  int h  = bh & 15;
  int q0 = qt * 32;

  const __bf16* Qp = Qb  + (size_t)bh * Sn * DHn;
  const __bf16* Kp = Kb  + (size_t)bh * Sn * DHn;
  const __bf16* Vp = VTb + (size_t)bh * DHn * Sn;
  const int*    Mp = maskp + (size_t)b * Sn * Sn;
  __bf16* pmem = &pTile[widb][0];

  // Q A-fragments [rowblk][dstep] (loop invariant)
  v16bf aq[2][2];
#pragma unroll
  for (int r = 0; r < 2; ++r) {
    aq[r][0] = ldA(Qp, DHn, q0 + r * 16, 0, lane);
    aq[r][1] = ldA(Qp, DHn, q0 + r * 16, 32, lane);
  }

  float Mx[2][8], L[2][8];
  v8f O[2][4] = {};
#pragma unroll
  for (int r = 0; r < 2; ++r)
#pragma unroll
    for (int j = 0; j < 8; ++j) { Mx[r][j] = -3.0e38f; L[r][j] = 0.0f; }

  for (int k0 = 0; k0 < Sn; k0 += 32) {
    // prefetch next-next chunk (K rows, V^T rows, mask rows); speculative, 1 line/lane
    int kpf = k0 + 64;
    if (kpf < Sn) {
      __builtin_prefetch(Kp + (size_t)(kpf + lane) * DHn, 0, 3);
      __builtin_prefetch(Vp + (size_t)lane * Sn + kpf, 0, 3);
      __builtin_prefetch(Vp + (size_t)(32 + lane) * Sn + kpf, 0, 3);
      __builtin_prefetch(Mp + (size_t)(q0 + lane) * Sn + kpf, 0, 3);
    }

    // K fragments [coltile][dstep] (shared by both row blocks) + V fragments
    v16bf bK[2][2];
#pragma unroll
    for (int t = 0; t < 2; ++t) {
      bK[t][0] = ldB(Kp, DHn, k0 + t * 16, 0, lane);
      bK[t][1] = ldB(Kp, DHn, k0 + t * 16, 32, lane);
    }
    v16bf bv[4];
#pragma unroll
    for (int i = 0; i < 4; ++i)
      bv[i] = __builtin_shufflevector(*(const v8bf*)(Vp + (size_t)(i * 16 + ln) * Sn + k0 + lh * 8),
                                      *(const v8bf*)(Vp + (size_t)(i * 16 + ln) * Sn + k0 + lh * 8 + 16),
                                      0,1,2,3,4,5,6,7,8,9,10,11,12,13,14,15);

    float al[2][8];
#pragma unroll
    for (int r = 0; r < 2; ++r) {
      // scores for this row block: two 16-col tiles over Dh=64
      v8f sc[2];
#pragma unroll
      for (int t = 0; t < 2; ++t) {
        v8f c = {};
        c = wmma_bf16(aq[r][0], bK[t][0], c);
        c = wmma_bf16(aq[r][1], bK[t][1], c);
        sc[t] = c;
      }
      // mask (before scaling, as in reference) then scale; online softmax
#pragma unroll
      for (int j = 0; j < 8; ++j) {
        int q = q0 + r * 16 + lh * 8 + j;
        int m0v = Mp[(size_t)q * Sn + k0 + ln];
        int m1v = Mp[(size_t)q * Sn + k0 + 16 + ln];
        float s0 = (m0v == 0) ? NEGS : sc[0][j] * SCL;
        float s1 = (m1v == 0) ? NEGS : sc[1][j] * SCL;
        float rm = fmaxf(s0, s1);
        rm = fmaxf(rm, __shfl_xor(rm, 1, 32));
        rm = fmaxf(rm, __shfl_xor(rm, 2, 32));
        rm = fmaxf(rm, __shfl_xor(rm, 4, 32));
        rm = fmaxf(rm, __shfl_xor(rm, 8, 32));
        float nm = fmaxf(Mx[r][j], rm);
        al[r][j] = __expf(Mx[r][j] - nm);
        Mx[r][j] = nm;
        float p0 = __expf(s0 - nm);
        float p1 = __expf(s1 - nm);
        float rs = p0 + p1;
        rs += __shfl_xor(rs, 1, 32);
        rs += __shfl_xor(rs, 2, 32);
        rs += __shfl_xor(rs, 4, 32);
        rs += __shfl_xor(rs, 8, 32);
        L[r][j] = L[r][j] * al[r][j] + rs;
        // stage P row-major [32][32] in per-wave LDS region
        pmem[(r * 16 + lh * 8 + j) * 32 + ln]      = f2bf(p0);
        pmem[(r * 16 + lh * 8 + j) * 32 + 16 + ln] = f2bf(p1);
      }
      // rescale O accumulators for this row block
#pragma unroll
      for (int i = 0; i < 4; ++i)
#pragma unroll
        for (int j = 0; j < 8; ++j) O[r][i][j] *= al[r][j];
    }

    // O += P(32x32) * V(32x64), per row block (per-wave LDS; DS pipe in-order)
#pragma unroll
    for (int r = 0; r < 2; ++r) {
      const __bf16* pr = pmem + (r * 16 + ln) * 32 + lh * 8;
      v8bf plo = *(const v8bf*)(pr);
      v8bf phi = *(const v8bf*)(pr + 16);
      v16bf ap = __builtin_shufflevector(plo, phi,
                   0,1,2,3,4,5,6,7,8,9,10,11,12,13,14,15);
#pragma unroll
      for (int i = 0; i < 4; ++i) O[r][i] = wmma_bf16(ap, bv[i], O[r][i]);
    }
  }

  // epilogue: ctx[b*S+q][h*64+d] = O/L  (bf16, feeds output projection)
#pragma unroll
  for (int r = 0; r < 2; ++r)
#pragma unroll
    for (int i = 0; i < 4; ++i) {
      int d = i * 16 + ln;
#pragma unroll
      for (int j = 0; j < 8; ++j) {
        int q = q0 + r * 16 + lh * 8 + j;
        ctx[(size_t)(b * Sn + q) * HID + h * DHn + d] = f2bf(O[r][i][j] / L[r][j]);
      }
    }
}

// ---------------- output projection ----------------
// waves: (8192/32)*(1024/64) = 4096
__global__ void out_proj(const __bf16* __restrict__ ctx, const __bf16* __restrict__ WoT,
                         const float* __restrict__ bo, float* __restrict__ out) {
  int wid  = (blockIdx.x * blockDim.x + threadIdx.x) >> 5;
  int lane = threadIdx.x & 31;
  int m0 = (wid / (HID / 64)) * 32;
  int n0 = (wid % (HID / 64)) * 64;

  v8f acc[8] = {};
  gemm32x64(ctx, WoT, m0, n0, lane, acc);

  int lh = lane >> 4, ln = lane & 15;
#pragma unroll
  for (int r = 0; r < 2; ++r)
#pragma unroll
    for (int i = 0; i < 4; ++i) {
      int n = n0 + i * 16 + ln;
      float bias = bo[n];
#pragma unroll
      for (int j = 0; j < 8; ++j) {
        int m = m0 + r * 16 + lh * 8 + j;
        out[(size_t)m * HID + n] = acc[r * 4 + i][j] + bias;
      }
    }
}

// ---------------- launch ----------------
extern "C" void kernel_launch(void* const* d_in, const int* in_sizes, int n_in,
                              void* d_out, int out_size, void* d_ws, size_t ws_size,
                              hipStream_t stream) {
  const float* x    = (const float*)d_in[0];
  const float* y    = (const float*)d_in[1];
  const float* Wq   = (const float*)d_in[2];
  const float* Wk   = (const float*)d_in[3];
  const float* Wv   = (const float*)d_in[4];
  const float* Wo   = (const float*)d_in[5];
  const float* bo   = (const float*)d_in[6];
  const int*   mask = (const int*)d_in[7];
  float* out = (float*)d_out;

  constexpr size_t NX = (size_t)Mrows * HID;        // 8388608
  constexpr size_t NW = (size_t)HID * HID;          // 1048576
  constexpr size_t NQ = (size_t)Bn * Hh * Sn * DHn; // 8388608

  __bf16* p = (__bf16*)d_ws;
  __bf16* xbf = p; p += NX;
  __bf16* ybf = p; p += NX;
  __bf16* WqT = p; p += NW;
  __bf16* WkT = p; p += NW;
  __bf16* WvT = p; p += NW;
  __bf16* WoT = p; p += NW;
  __bf16* Qb  = p; p += NQ;
  __bf16* Kb  = p; p += NQ;
  __bf16* VTb = p; p += NQ;
  __bf16* ctx = p; p += NX;

  // 1) fp32 -> bf16 conversions (+ coalesced weight transposes)
  cvt_copy8<<<(int)((NX / 8 + 255) / 256), 256, 0, stream>>>(
      (const float4*)x, (v8bf*)xbf, (int)(NX / 8));
  cvt_copy8<<<(int)((NX / 8 + 255) / 256), 256, 0, stream>>>(
      (const float4*)y, (v8bf*)ybf, (int)(NX / 8));
  dim3 trg(HID / 32, HID / 32);
  cvt_tr<<<trg, 256, 0, stream>>>(Wq, WqT, HID, HID);
  cvt_tr<<<trg, 256, 0, stream>>>(Wk, WkT, HID, HID);
  cvt_tr<<<trg, 256, 0, stream>>>(Wv, WvT, HID, HID);
  cvt_tr<<<trg, 256, 0, stream>>>(Wo, WoT, HID, HID);

  // 2) QKV projection: 12288 waves
  proj_qkv<<<1536, 256, 0, stream>>>(xbf, ybf, WqT, WkT, WvT, Qb, Kb, VTb);

  // 3) attention: 4096 waves
  attn<<<512, 256, 0, stream>>>(Qb, Kb, VTb, mask, ctx);

  // 4) output projection: 4096 waves
  out_proj<<<512, 256, 0, stream>>>(ctx, WoT, bo, out);

  (void)in_sizes; (void)n_in; (void)out_size; (void)ws_size;
}
